// TGN_81879256531422
// MI455X (gfx1250) — matrix operand order repaired and checked by reading, not compiled, not run on hardware
//
#include <hip/hip_runtime.h>
#include <hip/hip_bf16.h>
#include <cmath>

// ---------------------------------------------------------------------------
// MI455X (gfx1250): all GEMMs via v_wmma_f32_16x16x32_bf16 (fp32 accum).
// Stage-1 attention = split-K flash attention with double-buffered LDS fed by
// GLOBAL_LOAD_ASYNC_TO_LDS_B128 (ASYNCcnt) for the K blocks. wave32.
// ---------------------------------------------------------------------------

typedef __attribute__((ext_vector_type(16))) __bf16 v16bf;
typedef __attribute__((ext_vector_type(8)))  float  v8f;

#define LDQ 132   // padded f32 row stride for 129-wide buffers
#define LDB 168   // padded bf16 row stride (336 B, 16B aligned, bank-friendly)

__device__ __forceinline__ unsigned short f2bf(float f) {
  unsigned u = __float_as_uint(f);
  u += 0x7FFFu + ((u >> 16) & 1u);          // round-to-nearest-even
  return (unsigned short)(u >> 16);
}

union AF { v16bf v; unsigned u[8]; };

__device__ __forceinline__ void async_copy16(unsigned lds_off, const void* g) {
  asm volatile("global_load_async_to_lds_b128 %0, %1, off"
               :: "v"(lds_off), "v"(g) : "memory");
}
__device__ __forceinline__ void wait_async0() {
  asm volatile("s_wait_asynccnt 0" ::: "memory");
}

// ---------------- A-matrix gather descriptor -------------------------------
enum { AM_NBHD = 0, AM_QRY = 2, AM_PLAIN = 3, AM_CAT = 4 };

struct ASrc {
  const float* p0;
  const float* p1;
  const float* p2;
  int mode;
  int lda;
};

__device__ __forceinline__ float agather(const ASrc& s, int r, int j, int M, int Kact) {
  if (r >= M || j >= Kact) return 0.f;
  switch (s.mode) {
    case AM_NBHD: {                       // row r of [emb|ef|td] x5 concat (257 each)
      int c = j / 257; int w = j - c * 257;
      if (w < 128) return s.p0[(r * 5 + c) * 128 + w];
      if (w < 256) return s.p1[(r * 5 + c) * 128 + (w - 128)];
      return s.p2[r * 5 + c];
    }
    case AM_QRY: {                        // [emb(128) | td(1)]
      if (j < 128) return s.p0[r * 128 + j];
      return s.p2 ? s.p2[r] : 0.f;
    }
    case AM_CAT: {                        // [a(129, ld LDQ) | emb(128)]
      if (j < 129) return s.p0[(size_t)r * LDQ + j];
      return s.p1[r * 128 + (j - 129)];
    }
    default:                              // plain f32 row-major
      return s.p0[(size_t)r * s.lda + j];
  }
}

// ---------------- generic WMMA GEMM: out[b] = act((A @ W[b]^T + bias)*scale) -
template<int NTILES, int NB>
__global__ void __launch_bounds__(128)
gemm_wmma(ASrc a, int M, int Kact,
          const float* w0, const float* bias0, void* out0,
          const float* w1, const float* bias1, void* out1,
          int Nact, int ldout, float scale, int relu, int obf16) {
  __shared__ unsigned short Ash[64 * 40];
  __shared__ unsigned short Bsh[NB][NTILES * 16 * 40];

  const int tid  = threadIdx.x;
  const int wave = tid >> 5;
  const int lane = tid & 31;
  const int half = lane >> 4;
  const int nm   = lane & 15;
  const int r0   = blockIdx.x * 64;

  const float* W[NB]; const float* Bias[NB]; void* Out[NB];
  W[0] = w0; Bias[0] = bias0; Out[0] = out0;
  if (NB > 1) { W[1] = w1; Bias[1] = bias1; Out[1] = out1; }

  v8f acc[NB][NTILES];
  for (int b = 0; b < NB; ++b)
    for (int t = 0; t < NTILES; ++t)
      for (int e = 0; e < 8; ++e) acc[b][t][e] = 0.f;

  const int ksteps = (Kact + 31) >> 5;
  for (int ks = 0; ks < ksteps; ++ks) {
    const int k0 = ks << 5;
    for (int idx = tid; idx < 64 * 32; idx += 128) {
      int rr = idx >> 5, cc = idx & 31;
      Ash[rr * 40 + cc] = f2bf(agather(a, r0 + rr, k0 + cc, M, Kact));
    }
    for (int b = 0; b < NB; ++b)
      for (int idx = tid; idx < NTILES * 16 * 32; idx += 128) {
        int nn = idx >> 5, cc = idx & 31;
        float v = (nn < Nact && (k0 + cc) < Kact) ? W[b][nn * Kact + k0 + cc] : 0.f;
        Bsh[b][nn * 40 + cc] = f2bf(v);
      }
    __syncthreads();

    AF af;
#pragma unroll
    for (int v8 = 0; v8 < 8; ++v8) {
      int K0 = ((v8 >> 2) << 4) + (half << 3) + ((v8 & 3) << 1);
      af.u[v8] = *(const unsigned*)&Ash[(wave * 16 + nm) * 40 + K0];
    }
    for (int b = 0; b < NB; ++b) {
#pragma unroll
      for (int t = 0; t < NTILES; ++t) {
        AF bf;
#pragma unroll
        for (int v8 = 0; v8 < 8; ++v8) {
          int K0 = ((v8 >> 2) << 4) + (half << 3) + ((v8 & 3) << 1);
          bf.u[v8] = *(const unsigned*)&Bsh[b][(t * 16 + nm) * 40 + K0];
        }
        acc[b][t] = __builtin_amdgcn_wmma_f32_16x16x32_bf16(
            false, af.v, false, bf.v, (short)0, acc[b][t], false, false);
      }
    }
    __syncthreads();
  }

  for (int b = 0; b < NB; ++b) {
#pragma unroll
    for (int t = 0; t < NTILES; ++t) {
      int col = t * 16 + nm;
      if (col >= Nact) continue;
      float bias = Bias[b][col];
#pragma unroll
      for (int rr = 0; rr < 8; ++rr) {
        int row = r0 + wave * 16 + rr + half * 8;
        if (row >= M) continue;
        float v = (acc[b][t][rr] + bias) * scale;
        if (relu) v = fmaxf(v, 0.f);
        if (obf16) ((unsigned short*)Out[b])[(size_t)row * ldout + col] = f2bf(v);
        else       ((float*)Out[b])[(size_t)row * ldout + col] = v;
      }
    }
  }
}

// ---------------- stage-0 attention: 5 keys per query, pure VALU -----------
__global__ void __launch_bounds__(128)
attn0_kernel(const float* qp, const float* kp, const float* vp, float* out, int NQ) {
  int q = blockIdx.x * 4 + (threadIdx.x >> 5);
  int lane = threadIdx.x & 31;
  if (q >= NQ) return;
  int base = (q / 5) * 5;
  float s[5];
#pragma unroll
  for (int j = 0; j < 5; ++j) {
    float p = 0.f;
    for (int c = lane; c < 129; c += 32)
      p += qp[(size_t)q * LDQ + c] * kp[(size_t)(base + j) * LDQ + c];
    for (int m = 16; m >= 1; m >>= 1) p += __shfl_xor(p, m, 32);
    s[j] = p;
  }
  float mx = s[0];
  for (int j = 1; j < 5; ++j) mx = fmaxf(mx, s[j]);
  float sum = 0.f;
  for (int j = 0; j < 5; ++j) { s[j] = __expf(s[j] - mx); sum += s[j]; }
  float inv = 1.f / sum;
  for (int c = lane; c < 129; c += 32) {
    float o = 0.f;
#pragma unroll
    for (int j = 0; j < 5; ++j) o += s[j] * vp[(size_t)(base + j) * LDQ + c];
    out[(size_t)q * LDQ + c] = o * inv;
  }
}

// ---------------- zero-fill for padded bf16 buffers ------------------------
__global__ void zero_u16(unsigned short* p, int n) {
  int i = blockIdx.x * 256 + threadIdx.x;
  if (i < n) p[i] = 0;
}

// ---------------- stage-1 flash attention (split-K, async double-buffer) ---
// grid = (M/64) * KSPLIT; each WG: 4 waves x 16 queries over one key split.
// K blocks async-copied (raw bf16) into double-buffered LDS; V transposed by
// VALU staging; P converted C->A layout via wave-private LDS (in-order DS).
__global__ void __launch_bounds__(128)
attn1_kernel(const unsigned short* qpb, const unsigned short* kpb,
             const unsigned short* vpb,
             float* opart, float* mpart, float* lpart,
             int M, int NK, int qwgs, int nkps) {
  __shared__ unsigned short Ksh[2][32 * LDB];   // [key][feat0..167] raw rows
  __shared__ unsigned short Vsh[2][144 * 40];   // [feat][key0..31] transposed
  __shared__ unsigned short Psh[4 * 16 * 40];

  const int tid  = threadIdx.x;
  const int wave = tid >> 5, lane = tid & 31;
  const int half = lane >> 4, nm = lane & 15;
  const int qwg  = blockIdx.x % qwgs;
  const int sp   = blockIdx.x / qwgs;
  const int qrow0 = qwg * 64 + wave * 16;
  const int kbeg  = sp * nkps;

  // Q fragments (A-layout), direct from padded bf16 (cols>=129 are zero)
  AF qa[5];
#pragma unroll
  for (int kc = 0; kc < 5; ++kc)
#pragma unroll
    for (int v8 = 0; v8 < 8; ++v8) {
      int K0 = kc * 32 + ((v8 >> 2) << 4) + (half << 3) + ((v8 & 3) << 1);
      qa[kc].u[v8] = *(const unsigned*)&qpb[(size_t)(qrow0 + nm) * LDB + K0];
    }

  v8f o[9];
  for (int t = 0; t < 9; ++t)
    for (int e = 0; e < 8; ++e) o[t][e] = 0.f;
  float mrow[8], lrow[8];
  for (int r = 0; r < 8; ++r) { mrow[r] = -3.0e38f; lrow[r] = 0.f; }

  // preload block 0
  {
    const char* src = (const char*)(kpb + (size_t)kbeg * LDB);
    unsigned kb = (unsigned)(size_t)&Ksh[0][0];
    for (int idx = tid; idx < 672; idx += 128)          // 32*336B in 16B chunks
      async_copy16(kb + idx * 16, src + idx * 16);
    const unsigned short* vs = vpb + (size_t)kbeg * LDB;
    for (int idx = tid; idx < 576; idx += 128) {        // 32 keys x 18 oct-feats
      int k = idx / 18, c = idx - k * 18;
      uint4 d = *(const uint4*)(vs + (size_t)k * LDB + c * 8);
      const unsigned short* e = (const unsigned short*)&d;
#pragma unroll
      for (int e8 = 0; e8 < 8; ++e8) Vsh[0][(c * 8 + e8) * 40 + k] = e[e8];
    }
  }
  wait_async0();
  __syncthreads();

  int p = 0;
  const int nblk = nkps >> 5;
  for (int blk = 0; blk < nblk; ++blk) {
    const int j0 = kbeg + blk * 32;
    // prefetch next block into the other buffer (overlaps with WMMA below)
    if (blk + 1 < nblk) {
      const char* src = (const char*)(kpb + (size_t)(j0 + 32) * LDB);
      unsigned kb = (unsigned)(size_t)&Ksh[p ^ 1][0];
      for (int idx = tid; idx < 672; idx += 128)
        async_copy16(kb + idx * 16, src + idx * 16);
      const unsigned short* vs = vpb + (size_t)(j0 + 32) * LDB;
      for (int idx = tid; idx < 576; idx += 128) {
        int k = idx / 18, c = idx - k * 18;
        uint4 d = *(const uint4*)(vs + (size_t)k * LDB + c * 8);
        const unsigned short* e = (const unsigned short*)&d;
#pragma unroll
        for (int e8 = 0; e8 < 8; ++e8) Vsh[p ^ 1][(c * 8 + e8) * 40 + k] = e[e8];
      }
    }

    // S = Qp @ Kp^T : two 16x16 key tiles
    v8f s[2];
#pragma unroll
    for (int t = 0; t < 2; ++t) {
      for (int e = 0; e < 8; ++e) s[t][e] = 0.f;
#pragma unroll
      for (int kc = 0; kc < 5; ++kc) {
        AF bf;
#pragma unroll
        for (int v8 = 0; v8 < 8; ++v8) {
          int K0 = kc * 32 + ((v8 >> 2) << 4) + (half << 3) + ((v8 & 3) << 1);
          bf.u[v8] = *(const unsigned*)&Ksh[p][(t * 16 + nm) * LDB + K0];
        }
        s[t] = __builtin_amdgcn_wmma_f32_16x16x32_bf16(
            false, qa[kc].v, false, bf.v, (short)0, s[t], false, false);
      }
    }

    // online softmax (row = r + 8*half; 16 lanes of a half share row stats)
#pragma unroll
    for (int r = 0; r < 8; ++r) {
      float tm = fmaxf(s[0][r], s[1][r]);
      for (int m = 8; m >= 1; m >>= 1) tm = fmaxf(tm, __shfl_xor(tm, m, 32));
      float mnew = fmaxf(mrow[r], tm);
      float corr = __expf(mrow[r] - mnew);
      float p0 = __expf(s[0][r] - mnew);
      float p1 = __expf(s[1][r] - mnew);
      float rs = p0 + p1;
      for (int m = 8; m >= 1; m >>= 1) rs += __shfl_xor(rs, m, 32);
      lrow[r] = lrow[r] * corr + rs;
      mrow[r] = mnew;
#pragma unroll
      for (int t = 0; t < 9; ++t) o[t][r] *= corr;
      Psh[wave * 640 + (r + half * 8) * 40 + nm]      = f2bf(p0);
      Psh[wave * 640 + (r + half * 8) * 40 + 16 + nm] = f2bf(p1);
    }
    // same-wave LDS is executed in order: P stores land before these loads
    AF pf;
#pragma unroll
    for (int v8 = 0; v8 < 8; ++v8) {
      int K0 = ((v8 >> 2) << 4) + (half << 3) + ((v8 & 3) << 1);
      pf.u[v8] = *(const unsigned*)&Psh[wave * 640 + nm * 40 + K0];
    }
#pragma unroll
    for (int t = 0; t < 9; ++t) {
      AF vf;
#pragma unroll
      for (int v8 = 0; v8 < 8; ++v8) {
        int K0 = ((v8 >> 2) << 4) + (half << 3) + ((v8 & 3) << 1);
        vf.u[v8] = *(const unsigned*)&Vsh[p][(t * 16 + nm) * 40 + K0];
      }
      o[t] = __builtin_amdgcn_wmma_f32_16x16x32_bf16(
          false, pf.v, false, vf.v, (short)0, o[t], false, false);
    }

    wait_async0();
    __syncthreads();
    p ^= 1;
  }

  // write un-normalized partials + per-row (m, l)
#pragma unroll
  for (int t = 0; t < 9; ++t) {
    int col = t * 16 + nm;
    if (col >= 129) continue;
#pragma unroll
    for (int r = 0; r < 8; ++r) {
      int row = qrow0 + r + half * 8;
      opart[((size_t)sp * M + row) * LDQ + col] = o[t][r];
    }
  }
  if (nm == 0) {
#pragma unroll
    for (int r = 0; r < 8; ++r) {
      int row = qrow0 + r + half * 8;
      mpart[(size_t)sp * M + row] = mrow[r];
      lpart[(size_t)sp * M + row] = lrow[r];
    }
  }
}

__global__ void __launch_bounds__(256)
attn1_combine(const float* opart, const float* mpart, const float* lpart,
              float* out, int M, int S) {
  int idx = blockIdx.x * 256 + threadIdx.x;
  int q = idx / 129, c = idx - q * 129;
  if (q >= M) return;
  float Mx = -3.0e38f;
  for (int s = 0; s < S; ++s) Mx = fmaxf(Mx, mpart[s * M + q]);
  float L = 0.f, O = 0.f;
  for (int s = 0; s < S; ++s) {
    float w = __expf(mpart[s * M + q] - Mx);
    L += w * lpart[s * M + q];
    O += w * opart[((size_t)s * M + q) * LDQ + c];
  }
  out[(size_t)q * LDQ + c] = O / L;
}

// ---------------------------------------------------------------------------
extern "C" void kernel_launch(void* const* d_in, const int* in_sizes, int n_in,
                              void* d_out, int out_size, void* d_ws, size_t ws_size,
                              hipStream_t stream) {
  const float* emb0 = (const float*)d_in[0];
  const float* ef0  = (const float*)d_in[1];
  const float* td0  = (const float*)d_in[2];
  const float* emb1 = (const float*)d_in[3];
  const float* ef1  = (const float*)d_in[4];
  const float* td1  = (const float*)d_in[5];
  const float* emb2 = (const float*)d_in[6];
  const float* Wq = (const float*)d_in[7];  const float* bq = (const float*)d_in[8];
  const float* Wk = (const float*)d_in[9];  const float* bk = (const float*)d_in[10];
  const float* Wv = (const float*)d_in[11]; const float* bv = (const float*)d_in[12];
  const float* Wo = (const float*)d_in[13]; const float* bo = (const float*)d_in[14];
  const float* W1 = (const float*)d_in[15]; const float* b1 = (const float*)d_in[16];
  const float* W2 = (const float*)d_in[17]; const float* b2 = (const float*)d_in[18];

  const int B = 8192, M0 = B * 5;
  const int QD = 129, KD = 1285;
  const int KSPLIT = 4;
  const float scq = 1.0f / sqrtf(129.0f);

  float* ws = (float*)d_ws;
  const size_t P129 = (size_t)M0 * LDQ;
  const size_t S129 = (size_t)B * LDQ;

  // ---- stage-0 buffers (f32) ----
  float* kp0 = ws;             float* vp0 = ws + P129;
  float* qp0 = ws + 2 * P129;  float* o0  = ws + 3 * P129;
  float* a0  = kp0;            float* h0  = vp0;

  // ---- stage-1 buffers: bf16 q/k/v (contiguous, then f32 region) ----
  const size_t SBE = (size_t)B * LDB;                 // elems per bf16 buffer
  unsigned short* kp1b = (unsigned short*)d_ws;
  unsigned short* vp1b = kp1b + SBE;
  unsigned short* qp1b = kp1b + 2 * SBE;
  float* f1   = (float*)(kp1b + 3 * SBE);             // 16B-aligned (SBE*2 % 16 == 0)
  float* opart = f1;                                  // KSPLIT * B * LDQ
  float* mpart = opart + (size_t)KSPLIT * B * LDQ;    // KSPLIT * B
  float* lpart = mpart + (size_t)KSPLIT * B;
  float* o1    = lpart + (size_t)KSPLIT * B;
  float* a1    = o1 + S129;
  float* h1    = a1 + S129;

  ASrc s;
  dim3 blk(128);

  // ================= stage 0 =================
  s = {emb0, ef0, td0, AM_NBHD, 0};
  gemm_wmma<9, 2><<<(M0 + 63) / 64, blk, 0, stream>>>(
      s, M0, KD, Wk, bk, kp0, Wv, bv, vp0, 129, LDQ, 1.f, 0, 0);
  s = {emb1, nullptr, td1, AM_QRY, 0};
  gemm_wmma<9, 1><<<(M0 + 63) / 64, blk, 0, stream>>>(
      s, M0, QD, Wq, bq, qp0, nullptr, nullptr, nullptr, 129, LDQ, scq, 0, 0);
  attn0_kernel<<<(M0 + 3) / 4, blk, 0, stream>>>(qp0, kp0, vp0, o0, M0);
  s = {o0, nullptr, nullptr, AM_PLAIN, LDQ};
  gemm_wmma<9, 1><<<(M0 + 63) / 64, blk, 0, stream>>>(
      s, M0, QD, Wo, bo, a0, nullptr, nullptr, nullptr, 129, LDQ, 1.f, 0, 0);
  s = {a0, emb1, nullptr, AM_CAT, 0};
  gemm_wmma<8, 1><<<(M0 + 63) / 64, blk, 0, stream>>>(
      s, M0, 257, W1, b1, h0, nullptr, nullptr, nullptr, 128, 128, 1.f, 1, 0);
  s = {h0, nullptr, nullptr, AM_PLAIN, 128};
  gemm_wmma<8, 1><<<(M0 + 63) / 64, blk, 0, stream>>>(
      s, M0, 128, W2, b2, (float*)d_out, nullptr, nullptr, nullptr, 128, 128, 1.f, 0, 0);

  // ================= stage 1 =================
  // zero the padded bf16 q/k/v region (cols 129..167 must be 0 for WMMA K-pad)
  {
    int n = (int)(3 * SBE);
    zero_u16<<<(n + 255) / 256, 256, 0, stream>>>(kp1b, n);
  }
  s = {emb1, ef1, td1, AM_NBHD, 0};
  gemm_wmma<9, 2><<<(B + 63) / 64, blk, 0, stream>>>(
      s, B, KD, Wk, bk, kp1b, Wv, bv, vp1b, 129, LDB, 1.f, 0, 1);
  s = {emb2, nullptr, nullptr, AM_QRY, 0};
  gemm_wmma<9, 1><<<(B + 63) / 64, blk, 0, stream>>>(
      s, B, QD, Wq, bq, qp1b, nullptr, nullptr, nullptr, 129, LDB, scq, 0, 1);

  attn1_kernel<<<(B / 64) * KSPLIT, blk, 0, stream>>>(
      qp1b, kp1b, vp1b, opart, mpart, lpart, B, B, B / 64, B / KSPLIT);
  attn1_combine<<<(B * 129 + 255) / 256, 256, 0, stream>>>(
      opart, mpart, lpart, o1, B, KSPLIT);

  s = {o1, nullptr, nullptr, AM_PLAIN, LDQ};
  gemm_wmma<9, 1><<<(B + 63) / 64, blk, 0, stream>>>(
      s, B, QD, Wo, bo, a1, nullptr, nullptr, nullptr, 129, LDQ, 1.f, 0, 0);
  s = {a1, emb2, nullptr, AM_CAT, 0};
  gemm_wmma<8, 1><<<(B + 63) / 64, blk, 0, stream>>>(
      s, B, 257, W1, b1, h1, nullptr, nullptr, nullptr, 128, 128, 1.f, 1, 0);
  s = {h1, nullptr, nullptr, AM_PLAIN, 128};
  gemm_wmma<8, 1><<<(B + 63) / 64, blk, 0, stream>>>(
      s, B, 128, W2, b2, (float*)d_out + (size_t)M0 * 128, nullptr, nullptr, nullptr,
      128, 128, 1.f, 0, 0);
}
